// SparseSelfAttention_43026982371634
// MI455X (gfx1250) — matrix operand order repaired and checked by reading, not compiled, not run on hardware
//
#include <hip/hip_runtime.h>

// ---------------- problem constants ----------------
#define BB 4
#define TT 1344
#define DD 1024
#define HH 16
#define DHD 64
#define GG 84          // TT / 16
#define WW1 7
#define NKEY (GG + 2*WW1 + 1)   // 84 + 15 = 99
#define BT (BB*TT)              // 5376
#define NEGF (-3.402823466e38f)

typedef __attribute__((ext_vector_type(16))) _Float16 v16h;
typedef __attribute__((ext_vector_type(8)))  _Float16 v8h;
typedef __attribute__((ext_vector_type(8)))  float    v8f;

static __device__ __forceinline__ float wave_sum(float v) {
#pragma unroll
  for (int o = 16; o > 0; o >>= 1) v += __shfl_xor(v, o, 32);
  return v;
}
static __device__ __forceinline__ float wave_max(float v) {
#pragma unroll
  for (int o = 16; o > 0; o >>= 1) v = fmaxf(v, __shfl_xor(v, o, 32));
  return v;
}
static __device__ __forceinline__ int imin(int a, int b) { return a < b ? a : b; }
static __device__ __forceinline__ int imax(int a, int b) { return a > b ? a : b; }

// ---------------- WMMA fragment loaders ----------------
// A (16x32 f16, row major, leading dim ld): lane L holds row (L&15);
// elements 0..7 = K[c0..c0+7], 8..15 = K[16+c0..16+c0+7], c0 = (L>>4)*8.
static __device__ __forceinline__ v16h load_fragA(const _Float16* base, int ld, int lane) {
  int r  = lane & 15;
  int c0 = (lane >> 4) * 8;
  const _Float16* p = base + (size_t)r * ld + c0;
  v8h lo = *(const v8h*)(p);
  v8h hi = *(const v8h*)(p + 16);
  v16h f;
#pragma unroll
  for (int i = 0; i < 8; i++) { f[i] = lo[i]; f[8 + i] = hi[i]; }
  return f;
}
// B (32x16 f16). Source is W stored [N][K] row major, which is exactly the
// B-matrix column for output n. Lane L (n = L&15) holds elements 0..7 =
// K[c0..c0+7], 8..15 = K[16+c0..16+c0+7], c0 = (L>>4)*8 (same as A).
static __device__ __forceinline__ v16h load_fragB(const _Float16* base, int ld, int lane) {
  int n  = lane & 15;
  int h  = lane >> 4;
  const _Float16* p = base + (size_t)n * ld + h * 8;
  v8h lo = *(const v8h*)(p);
  v8h hi = *(const v8h*)(p + 16);
  v16h f;
#pragma unroll
  for (int i = 0; i < 8; i++) { f[i] = lo[i]; f[8 + i] = hi[i]; }
  return f;
}
static __device__ __forceinline__ v8f wmma32(v16h a, v16h b, v8f c) {
  return __builtin_amdgcn_wmma_f32_16x16x32_f16(false, a, false, b, (short)0, c, false, false);
}

// ---------------- kernels ----------------

// LayerNorm f32 -> f16, one wave per token (8 waves / block)
__global__ void ssa_layernorm(const float* __restrict__ x,
                              const float* __restrict__ g,
                              const float* __restrict__ b,
                              _Float16* __restrict__ xn) {
  int lane = threadIdx.x & 31;
  int tok  = blockIdx.x * 8 + (threadIdx.x >> 5);
  if (tok >= BT) return;
  const float* xp = x + (size_t)tok * DD;
  float xv[32];
  float s = 0.f, s2 = 0.f;
#pragma unroll
  for (int i = 0; i < 32; i++) {
    float v = xp[i * 32 + lane];
    xv[i] = v; s += v; s2 += v * v;
  }
  s  = wave_sum(s);
  s2 = wave_sum(s2);
  float mu  = s * (1.f / DD);
  float var = s2 * (1.f / DD) - mu * mu;
  float r   = rsqrtf(var + 1e-5f);
  _Float16* op = xn + (size_t)tok * DD;
#pragma unroll
  for (int i = 0; i < 32; i++) {
    int d = i * 32 + lane;
    op[d] = (_Float16)((xv[i] - mu) * r * g[d] + b[d]);
  }
}

__global__ void ssa_cvt_f16(const float* __restrict__ s, _Float16* __restrict__ d, int n) {
  int i = blockIdx.x * blockDim.x + threadIdx.x;
  if (i < n) d[i] = (_Float16)s[i];
}

// C[M,N] = (A[M,K] @ W[N,K]^T + bias) * scale.  M=BT, N=K=1024.
// One wave -> 32x64 tile: 2 A frags x 4 B frags -> 8 WMMAs per k-step.
__global__ void ssa_gemm(const _Float16* __restrict__ A,
                         const _Float16* __restrict__ W,
                         const float* __restrict__ bias,
                         float* __restrict__ C, float scale) {
  const int K = DD, N = DD;
  int lane = threadIdx.x & 31;
  int wave = blockIdx.x * (blockDim.x >> 5) + (threadIdx.x >> 5);
  const int tilesN = N / 64;                 // 16
  int tm = wave / tilesN;                    // 32-row tiles: 0..167
  int tn = wave % tilesN;
  if (tm >= BT / 32) return;
  v8f acc[2][4];
#pragma unroll
  for (int i = 0; i < 2; i++)
#pragma unroll
    for (int j = 0; j < 4; j++) acc[i][j] = (v8f){0.f,0.f,0.f,0.f,0.f,0.f,0.f,0.f};

  const _Float16* Abase = A + (size_t)tm * 32 * K;
  const _Float16* Wbase = W + (size_t)tn * 64 * K;
  for (int k0 = 0; k0 < K; k0 += 32) {
    v16h a0 = load_fragA(Abase + k0, K, lane);
    v16h a1 = load_fragA(Abase + (size_t)16 * K + k0, K, lane);
#pragma unroll
    for (int j = 0; j < 4; j++) {
      v16h bf = load_fragB(Wbase + (size_t)j * 16 * K + k0, K, lane);
      acc[0][j] = wmma32(a0, bf, acc[0][j]);
      acc[1][j] = wmma32(a1, bf, acc[1][j]);
    }
  }
  int nl   = lane & 15;
  int mrow = (lane >> 4) * 8;
#pragma unroll
  for (int i = 0; i < 2; i++) {
#pragma unroll
    for (int j = 0; j < 4; j++) {
      int n = tn * 64 + j * 16 + nl;
      float bv = bias[n];
      size_t base = (size_t)(tm * 32 + i * 16 + mrow) * N + n;
#pragma unroll
      for (int r = 0; r < 8; r++)
        C[base + (size_t)r * N] = (acc[i][j][r] + bv) * scale;
    }
  }
}

// Stable selection: global keyframes by t, then the rest by t; first G.
__global__ void ssa_gidx(const unsigned char* __restrict__ keyframe,
                         const unsigned char* __restrict__ valid,
                         int* __restrict__ gidx) {
  int b = threadIdx.x;
  if (b >= BB) return;
  int c = 0;
  for (int t = 0; t < TT && c < GG; t++)
    if (keyframe[b * TT + t] && valid[b * TT + t]) gidx[b * GG + c++] = t;
  for (int t = 0; t < TT && c < GG; t++)
    if (!(keyframe[b * TT + t] && valid[b * TT + t])) gidx[b * GG + c++] = t;
}

// Band + global-column attention: one wave per (b,t,h).
__global__ void ssa_band_attn(const float* __restrict__ q,
                              const float* __restrict__ k,
                              const float* __restrict__ v,
                              const int* __restrict__ gidx,
                              const unsigned char* __restrict__ keyframe,
                              const unsigned char* __restrict__ valid,
                              float* __restrict__ attn) {
  __shared__ float sq[8][DHD];
  __shared__ float sp[8][NKEY + 1];
  int wv   = threadIdx.x >> 5;
  int lane = threadIdx.x & 31;
  long gw  = (long)blockIdx.x * 8 + wv;
  int h = (int)(gw % HH);
  long bt = gw / HH;
  int t = (int)(bt % TT);
  int b = (int)(bt / TT);
  if (b >= BB) return;

  const float* qp = q + ((size_t)(b * TT + t) * DD) + h * DHD;
  sq[wv][lane]      = qp[lane];
  sq[wv][lane + 32] = qp[lane + 32];
  const int* gb = gidx + b * GG;

  // scores: lane covers keys lane, lane+32, lane+64, lane+96
  float sc[4];
#pragma unroll
  for (int s = 0; s < 4; s++) {
    int key = lane + 32 * s;
    float val = NEGF;
    if (key < NKEY) {
      int tok; bool masked = false;
      if (key < GG) {
        tok = gb[key];                       // global column: unmasked
      } else {
        int tt = t + (key - GG) - WW1;
        int ttc = imin(imax(tt, 0), TT - 1);
        bool inr = (tt >= 0) && (tt < TT);
        unsigned char vv = valid[b * TT + ttc];
        unsigned char kf = keyframe[b * TT + ttc];
        masked = (!inr) || (!vv) || (kf != 0);
        tok = ttc;
      }
      const float* kp = k + ((size_t)(b * TT + tok) * DD) + h * DHD;
      float dot = 0.f;
#pragma unroll 8
      for (int d = 0; d < DHD; d++) dot += sq[wv][d] * kp[d];
      val = masked ? NEGF : dot;
    }
    sc[s] = val;
  }
  float m = fmaxf(fmaxf(sc[0], sc[1]), fmaxf(sc[2], sc[3]));
  m = wave_max(m);
  float e[4];
  float ssum = 0.f;
#pragma unroll
  for (int s = 0; s < 4; s++) {
    int key = lane + 32 * s;
    e[s] = (key < NKEY) ? __expf(sc[s] - m) : 0.f;
    ssum += e[s];
  }
  ssum = wave_sum(ssum);
  float vmask = valid[b * TT + t] ? 1.f : 0.f;
  float inv = vmask / ssum;
#pragma unroll
  for (int s = 0; s < 4; s++) {
    int key = lane + 32 * s;
    if (key < NKEY) sp[wv][key] = e[s] * inv;
  }

  // output: lane covers dims lane, lane+32
  float o0 = 0.f, o1 = 0.f;
  for (int key = 0; key < NKEY; key++) {
    int tok;
    if (key < GG) tok = gb[key];
    else          tok = imin(imax(t + (key - GG) - WW1, 0), TT - 1);
    float p = sp[wv][key];
    const float* vp = v + ((size_t)(b * TT + tok) * DD) + h * DHD;
    o0 += p * vp[lane];
    o1 += p * vp[lane + 32];
  }
  float* op = attn + ((size_t)(b * TT + t) * DD) + h * DHD;
  op[lane]      = o0;
  op[lane + 32] = o1;
}

// Global attention: one wave per (b,h,g), 4 waves / block.
__global__ void ssa_global_attn(const float* __restrict__ qg,
                                const float* __restrict__ kg,
                                const float* __restrict__ vg,
                                const int* __restrict__ gidx,
                                const unsigned char* __restrict__ valid,
                                float* __restrict__ attn) {
  __shared__ float scl[4][TT];
  __shared__ float sq[4][DHD];
  int wv   = threadIdx.x >> 5;
  int lane = threadIdx.x & 31;
  long gw  = (long)blockIdx.x * 4 + wv;
  int g = (int)(gw % GG);
  int h = (int)((gw / GG) % HH);
  int b = (int)(gw / ((long)GG * HH));
  if (b >= BB) return;

  int tg = gidx[b * GG + g];
  const float* qp = qg + ((size_t)(b * TT + tg) * DD) + h * DHD;
  sq[wv][lane]      = qp[lane];
  sq[wv][lane + 32] = qp[lane + 32];

  float mloc = NEGF;
  for (int t = lane; t < TT; t += 32) {
    const float* kp = kg + ((size_t)(b * TT + t) * DD) + h * DHD;
    float dot = 0.f;
#pragma unroll 8
    for (int d = 0; d < DHD; d++) dot += sq[wv][d] * kp[d];
    float s = valid[b * TT + t] ? dot : NEGF;
    scl[wv][t] = s;
    mloc = fmaxf(mloc, s);
  }
  float m = wave_max(mloc);
  float ssum = 0.f;
  for (int t = lane; t < TT; t += 32) {
    float e = __expf(scl[wv][t] - m);
    scl[wv][t] = e;
    ssum += e;
  }
  ssum = wave_sum(ssum);
  float inv = 1.f / ssum;

  float o0 = 0.f, o1 = 0.f;
  for (int t = 0; t < TT; t++) {
    float p = scl[wv][t] * inv;
    const float* vp = vg + ((size_t)(b * TT + t) * DD) + h * DHD;
    o0 += p * vp[lane];
    o1 += p * vp[lane + 32];
  }
  float* op = attn + ((size_t)(b * TT + tg) * DD) + h * DHD;
  op[lane]      = o0;
  op[lane + 32] = o1;
}

__global__ void ssa_residual_add(const float* __restrict__ x,
                                 const float* __restrict__ attn,
                                 float* __restrict__ out, int n4) {
  int i = blockIdx.x * blockDim.x + threadIdx.x;
  if (i < n4) {
    float4 a = ((const float4*)x)[i];
    float4 c = ((const float4*)attn)[i];
    ((float4*)out)[i] = make_float4(a.x + c.x, a.y + c.y, a.z + c.z, a.w + c.w);
  }
}

// ---------------- host launcher ----------------
extern "C" void kernel_launch(void* const* d_in, const int* in_sizes, int n_in,
                              void* d_out, int out_size, void* d_ws, size_t ws_size,
                              hipStream_t stream) {
  const float* x            = (const float*)d_in[0];
  const unsigned char* vmsk = (const unsigned char*)d_in[1];  // src_mask (B,T,1) bool
  const unsigned char* kfrm = (const unsigned char*)d_in[2];  // keyframe_mask (B,T) bool
  const float* Wmat[6] = { (const float*)d_in[3], (const float*)d_in[5],
                           (const float*)d_in[7], (const float*)d_in[9],
                           (const float*)d_in[11], (const float*)d_in[13] };
  const float* bvec[6] = { (const float*)d_in[4], (const float*)d_in[6],
                           (const float*)d_in[8], (const float*)d_in[10],
                           (const float*)d_in[12], (const float*)d_in[14] };
  const float* ln_g = (const float*)d_in[15];
  const float* ln_b = (const float*)d_in[16];
  float* out = (float*)d_out;

  // workspace carve (reuse the 3 projection buffers between local & global phases)
  char* p = (char*)d_ws;
  auto carve = [&](size_t bytes) -> void* {
    void* r = (void*)p;
    p += (bytes + 255) & ~(size_t)255;
    return r;
  };
  _Float16* xnh  = (_Float16*)carve((size_t)BT * DD * 2);
  _Float16* wh   = (_Float16*)carve((size_t)DD * DD * 2);
  float*    p0   = (float*)carve((size_t)BT * DD * 4);   // q  / qg
  float*    p1   = (float*)carve((size_t)BT * DD * 4);   // k  / kg
  float*    p2   = (float*)carve((size_t)BT * DD * 4);   // v  / vg
  float*    attn = (float*)carve((size_t)BT * DD * 4);
  int*      gidx = (int*)carve((size_t)BB * GG * 4);
  (void)ws_size; (void)in_sizes; (void)n_in; (void)out_size;

  const float inv_sqrt_dh = 0.125f;   // 1/sqrt(64)

  // 1) LayerNorm -> f16
  ssa_layernorm<<<BT / 8, 256, 0, stream>>>(x, ln_g, ln_b, xnh);
  // 2) g_idx
  ssa_gidx<<<1, 32, 0, stream>>>(kfrm, vmsk, gidx);

  const int gemm_blocks = (BT / 32) * (DD / 64) / 8;  // 2688 waves / 8 = 336
  float* projs[3] = { p0, p1, p2 };

  // 3) local q,k,v projections + band attention
  for (int i = 0; i < 3; i++) {
    ssa_cvt_f16<<<(DD * DD) / 512, 512, 0, stream>>>(Wmat[i], wh, DD * DD);
    ssa_gemm<<<gemm_blocks, 256, 0, stream>>>(xnh, wh, bvec[i], projs[i],
                                              i == 0 ? inv_sqrt_dh : 1.0f);
  }
  ssa_band_attn<<<(BB * TT * HH) / 8, 256, 0, stream>>>(p0, p1, p2, gidx, kfrm, vmsk, attn);

  // 4) global qg,kg,vg projections (reuse buffers) + global attention
  for (int i = 0; i < 3; i++) {
    ssa_cvt_f16<<<(DD * DD) / 512, 512, 0, stream>>>(Wmat[3 + i], wh, DD * DD);
    ssa_gemm<<<gemm_blocks, 256, 0, stream>>>(xnh, wh, bvec[3 + i], projs[i],
                                              i == 0 ? inv_sqrt_dh : 1.0f);
  }
  ssa_global_attn<<<(BB * HH * GG) / 4, 128, 0, stream>>>(p0, p1, p2, gidx, vmsk, attn);

  // 5) residual add
  ssa_residual_add<<<(BT * DD / 4 + 255) / 256, 256, 0, stream>>>(x, attn, out, BT * DD / 4);
}
